// RNN_22694607192360
// MI455X (gfx1250) — matrix operand order repaired and checked by reading, not compiled, not run on hardware
//
#include <hip/hip_runtime.h>
#include <stdint.h>

#define B_ 64
#define T_ 512
#define D_ 256
#define H_ 1024
#define V_ 32000

typedef __attribute__((ext_vector_type(16))) __bf16        v16bf;
typedef __attribute__((ext_vector_type(8)))  float         v8f;
typedef __attribute__((ext_vector_type(4)))  unsigned int  v4u;
typedef __attribute__((ext_vector_type(4)))  unsigned int  u32x4;
typedef __attribute__((ext_vector_type(8)))  int           i32x8;
typedef __attribute__((ext_vector_type(4)))  int           i32x4;

union Frag {
    v16bf bf;
    v4u   q[2];
};

// round-to-nearest-even f32 -> bf16 (bit pattern)
static __device__ __forceinline__ unsigned short f2bf(float f) {
    union { float f; uint32_t u; } v; v.f = f;
    uint32_t u = v.u;
    return (unsigned short)((u + 0x7FFFu + ((u >> 16) & 1u)) >> 16);
}

// A fragment 16x32 bf16, row-major [M,K]: lane l -> row l%16,
// K = kb+0..15 contiguous, kb = (l<16 ? 0 : 16)
static __device__ __forceinline__ v16bf mk_a(const unsigned short* p) {
    Frag f;
    f.q[0] = *(const v4u*)(p);
    f.q[1] = *(const v4u*)(p + 8);
    return f.bf;
}

// B fragment 32x16 bf16 from BT stored [N,K]: lane l -> col l%16,
// K = kb+0..15 contiguous, kb = (l<16 ? 0 : 16)
static __device__ __forceinline__ v16bf mk_b(const unsigned short* p) {
    Frag f;
    f.q[0] = *(const v4u*)(p);
    f.q[1] = *(const v4u*)(p + 8);
    return f.bf;
}

// One 16x16 f32 tile of A[M,K] x BT[N,K]^T, K multiple of 64, all-global operands.
static __device__ __forceinline__ v8f wave_gemm(
    const unsigned short* __restrict__ A, int lda,
    const unsigned short* __restrict__ BT, int ldb,
    int m0, int n0, int K, int lane)
{
    v8f c0 = {};
    v8f c1 = {};
    const unsigned short* arow = A  + (size_t)(m0 + (lane & 15)) * lda + ((lane >> 4) << 4);
    const unsigned short* brow = BT + (size_t)(n0 + (lane & 15)) * ldb + ((lane >> 4) << 4);
    for (int k = 0; k < K; k += 64) {
        v16bf a0 = mk_a(arow + k);
        v16bf b0 = mk_b(brow + k);
        c0 = __builtin_amdgcn_wmma_f32_16x16x32_bf16(false, a0, false, b0,
                                                     (short)0, c0, false, false);
        v16bf a1 = mk_a(arow + k + 32);
        v16bf b1 = mk_b(brow + k + 32);
        c1 = __builtin_amdgcn_wmma_f32_16x16x32_bf16(false, a1, false, b1,
                                                     (short)0, c1, false, false);
    }
    return c0 + c1;
}

// ---------------- prep kernels ----------------

__global__ __launch_bounds__(256) void f32_to_bf16_kernel(
    const float* __restrict__ in, unsigned short* __restrict__ out, int n)
{
    int i = blockIdx.x * 256 + threadIdx.x;
    if (i < n) out[i] = f2bf(in[i]);
}

// in: [K,N] f32 row-major -> out: [N,K] bf16 row-major
__global__ __launch_bounds__(256) void transpose_to_bf16_kernel(
    const float* __restrict__ in, unsigned short* __restrict__ out, int K, int N)
{
    int idx = blockIdx.x * 256 + threadIdx.x;
    if (idx >= K * N) return;
    int nn = idx / K;
    int kk = idx - nn * K;
    out[idx] = f2bf(in[(size_t)kk * N + nn]);
}

__global__ __launch_bounds__(256) void zero_u16_kernel(unsigned short* p, int n)
{
    int i = blockIdx.x * 256 + threadIdx.x;
    if (i < n) p[i] = 0;
}

__global__ void zero_u32_kernel(unsigned int* p)
{
    if (threadIdx.x == 0 && blockIdx.x == 0) *p = 0u;
}

// ---------------- GEMM kernels ----------------

// xw[t,b,:] = x[b,t,:] @ W + bias   (M = B*T, K = D, N = H)
__global__ __launch_bounds__(128) void proj_gemm_kernel(
    const unsigned short* __restrict__ xb,   // [B*T, D] bf16
    const unsigned short* __restrict__ Wt,   // [H, D]  bf16 (transposed)
    const float* __restrict__ bias,          // [H]
    float* __restrict__ xw)                  // [T, B, H] f32
{
    int lane = threadIdx.x & 31;
    int wid  = (blockIdx.x << 2) + (threadIdx.x >> 5);
    int tile_m = wid >> 6;            // 0..2047
    int tile_n = wid & 63;            // 0..63
    int m0 = tile_m << 4, n0 = tile_n << 4;
    v8f acc = wave_gemm(xb, D_, Wt, D_, m0, n0, D_, lane);
    int n    = n0 + (lane & 15);
    int mrow = m0 + ((lane >> 4) << 3);
    float bv = bias[n];
#pragma unroll
    for (int r = 0; r < 8; r++) {
        int m  = mrow + r;
        int bb = m >> 9;              // m / T
        int t  = m & (T_ - 1);        // m % T
        xw[((size_t)t * B_ + bb) * H_ + n] = acc[r] + bv;
    }
}

// LDS staging geometry for the U tile (16 rows x 1024 bf16, TDM-padded):
//   pad_interval = 256 DWORDs (512 bf16), pad_amount = 2 DWORDs (4 bf16)
//   -> physical row stride = 1024 + 8 bf16 = 1032 elems = 516 DWORDs
//   516 % 64 == 4 banks/row => 16-lane ds_load_b128 is bank-conflict-free.
#define UROW_ 1032

// Persistent recurrence scan: h_{t+1} = tanh(xw_t + h_t @ U), t = 0..T-1.
// 64 resident WGs (one per 16-col U block). U block is TDM-loaded into LDS
// ONCE and reused for all 512 steps. Device-wide step barrier via a
// monotonically increasing global atomic counter.
__global__ __launch_bounds__(128) void rnn_scan_kernel(
    unsigned short* __restrict__ hA,          // [B,H] bf16, input at even t
    unsigned short* __restrict__ hB,          // [B,H] bf16, input at odd t
    const unsigned short* __restrict__ Ut,    // [H,H] bf16 (transposed, [N,K])
    const float* __restrict__ xw,             // [T,B,H] f32
    unsigned int* __restrict__ cnt)           // step barrier counter (zeroed)
{
    __shared__ unsigned short uTile[16 * UROW_];   // 33,024 B
    const int lane = threadIdx.x & 31;
    const int wid  = threadIdx.x >> 5;             // tile_m, 0..3
    const int n0   = blockIdx.x << 4;              // tile_n * 16

#if __has_builtin(__builtin_amdgcn_tensor_load_to_lds) && __has_builtin(__builtin_amdgcn_s_wait_tensorcnt)
    if (wid == 0) {
        // Tensor DMA descriptor (CDNA5 ISA): 2D tile of BT,
        // rows n0..n0+15, cols K=0..1023, 2-byte elements, LDS padding on.
        unsigned long long ga = (unsigned long long)(uintptr_t)(Ut + (size_t)n0 * H_);
        unsigned int lds = (unsigned int)(uintptr_t)&uTile[0];
        u32x4 g0;
        g0[0] = 1u;                                   // count=1 (valid), user mode
        g0[1] = lds;                                  // lds_addr (bytes)
        g0[2] = (unsigned int)ga;                     // global_addr[31:0]
        g0[3] = (unsigned int)((ga >> 32) & 0x1FFFFFFu) | (2u << 30); // addr[56:32] | type=2
        i32x8 g1;
        g1[0] = (int)0x03D10000u;   // data_size=2B | pad_enable | pad_interval=7 | pad_amount=1
        g1[1] = (int)(1024u << 16); // tensor_dim0 = 1024 (bits 79:48 low half)
        g1[2] = (int)(1024u << 16); // tensor_dim0 hi=0, tensor_dim1 = 1024
        g1[3] = (int)(1024u << 16); // tensor_dim1 hi=0, tile_dim0 = 1024
        g1[4] = 16;                 // tile_dim1 = 16, tile_dim2 = 0
        g1[5] = 1024;               // tensor_dim0_stride = 1024
        g1[6] = 0;                  // stride hi / tensor_dim1_stride lo
        g1[7] = 0;
        i32x4 z4 = (i32x4)(0);
#if defined(__clang_major__) && (__clang_major__ >= 23)
        i32x8 z8 = (i32x8)(0);
        __builtin_amdgcn_tensor_load_to_lds(g0, g1, z4, z4, z8, 0);
#else
        __builtin_amdgcn_tensor_load_to_lds(g0, g1, z4, z4, 0);
#endif
        __builtin_amdgcn_s_wait_tensorcnt(0);
    }
#else
    // Fallback: cooperative load with the same padded layout.
    for (int cc = threadIdx.x; cc < 16 * 128; cc += 128) {
        int row = cc >> 7;
        int c   = (cc & 127) << 3;
        int phys = row * UROW_ + c + ((c >> 9) << 2);
        *(v4u*)&uTile[phys] = *(const v4u*)(Ut + (size_t)(n0 + row) * H_ + c);
    }
#endif
    __syncthreads();

    const int    m0   = wid << 4;
    const int    brow = (lane & 15) * UROW_;
    const int    bkb  = (lane >> 4) << 4;
    const size_t aoff = (size_t)(m0 + (lane & 15)) * H_ + ((lane >> 4) << 4);
    const int    n    = n0 + (lane & 15);
    const int    mrow = m0 + ((lane >> 4) << 3);

    for (int t = 0; t < T_; ++t) {
        const unsigned short* h_in  = (t & 1) ? hB : hA;
        unsigned short*       h_out = (t & 1) ? hA : hB;
        const float* xw_t = xw + (size_t)t * (B_ * H_);

        v8f c0 = {};
        v8f c1 = {};
        const unsigned short* arow = h_in + aoff;
        for (int k = 0; k < H_; k += 64) {
            int kk0 = k + bkb;
            int kk1 = k + 32 + bkb;
            v16bf a0 = mk_a(arow + k);
            v16bf b0 = mk_b(&uTile[brow + kk0 + ((kk0 >> 9) << 2)]);
            c0 = __builtin_amdgcn_wmma_f32_16x16x32_bf16(false, a0, false, b0,
                                                         (short)0, c0, false, false);
            v16bf a1 = mk_a(arow + k + 32);
            v16bf b1 = mk_b(&uTile[brow + kk1 + ((kk1 >> 9) << 2)]);
            c1 = __builtin_amdgcn_wmma_f32_16x16x32_bf16(false, a1, false, b1,
                                                         (short)0, c1, false, false);
        }
        v8f acc = c0 + c1;

#pragma unroll
        for (int r = 0; r < 8; r++) {
            int m   = mrow + r;
            float v = acc[r] + xw_t[(size_t)m * H_ + n];
            h_out[(size_t)m * H_ + n] = f2bf(tanhf(v));
        }

        // ---- device-wide step barrier ----
        __threadfence();              // make this block's h stores visible
        __syncthreads();              // all waves of this WG arrived
        if (threadIdx.x == 0) {
            atomicAdd(cnt, 1u);
            const unsigned int target = 64u * (unsigned int)(t + 1);
            while (*(volatile const unsigned int*)cnt < target) {
                __builtin_amdgcn_s_sleep(1);
            }
        }
        __syncthreads();              // release: everyone sees step t done
        __threadfence();
    }
}

// logits = hT @ Wd + bd   (M = B, K = H, N = V)
__global__ __launch_bounds__(128) void logits_gemm_kernel(
    const unsigned short* __restrict__ h,     // [B, H] bf16
    const unsigned short* __restrict__ Wdt,   // [V, H] bf16 (transposed)
    const float* __restrict__ bd,             // [V]
    float* __restrict__ logits)               // [B, V]
{
    int lane = threadIdx.x & 31;
    int wid  = (blockIdx.x << 2) + (threadIdx.x >> 5);   // 0..7999
    int tile_m = wid / 2000;          // 0..3
    int tile_n = wid - tile_m * 2000; // 0..1999
    int m0 = tile_m << 4, n0 = tile_n << 4;
    v8f acc = wave_gemm(h, H_, Wdt, H_, m0, n0, H_, lane);
    int n    = n0 + (lane & 15);
    int mrow = m0 + ((lane >> 4) << 3);
    float bv = bd[n];
#pragma unroll
    for (int r = 0; r < 8; r++) {
        int m = mrow + r;
        logits[(size_t)m * V_ + n] = acc[r] + bv;
    }
}

// ---------------- softmax ----------------

__global__ __launch_bounds__(256) void softmax_kernel(
    const float* __restrict__ logits, float* __restrict__ out)
{
    const int row = blockIdx.x;
    const float* p = logits + (size_t)row * V_;
    float* o = out + (size_t)row * V_;
    __shared__ float sred[8];
    int tid = threadIdx.x, lane = tid & 31, wid = tid >> 5;

    float mx = -3.402823e38f;
    for (int i = tid; i < V_; i += 256) mx = fmaxf(mx, p[i]);
    for (int off = 16; off; off >>= 1) mx = fmaxf(mx, __shfl_xor(mx, off, 32));
    if (lane == 0) sred[wid] = mx;
    __syncthreads();
    float m0 = sred[0];
#pragma unroll
    for (int i = 1; i < 8; i++) m0 = fmaxf(m0, sred[i]);
    __syncthreads();

    float sum = 0.f;
    for (int i = tid; i < V_; i += 256) sum += __expf(p[i] - m0);
    for (int off = 16; off; off >>= 1) sum += __shfl_xor(sum, off, 32);
    if (lane == 0) sred[wid] = sum;
    __syncthreads();
    float s0 = 0.f;
#pragma unroll
    for (int i = 0; i < 8; i++) s0 += sred[i];
    float inv = 1.0f / s0;

    for (int i = tid; i < V_; i += 256) o[i] = __expf(p[i] - m0) * inv;
}

// ---------------- launch ----------------

extern "C" void kernel_launch(void* const* d_in, const int* in_sizes, int n_in,
                              void* d_out, int out_size, void* d_ws, size_t ws_size,
                              hipStream_t stream) {
    const float* x  = (const float*)d_in[0];   // [B,T,D]
    const float* W  = (const float*)d_in[1];   // [D,H]
    const float* U  = (const float*)d_in[2];   // [H,H]
    const float* bv = (const float*)d_in[3];   // [H]
    const float* Wd = (const float*)d_in[4];   // [H,V]
    const float* bd = (const float*)d_in[5];   // [V]
    float* out = (float*)d_out;                // [B,V]

    size_t off = 0;
    char* base = (char*)d_ws;
    auto alloc = [&](size_t bytes) -> void* {
        void* p = base + off;
        off += (bytes + 255) & ~(size_t)255;
        return p;
    };
    unsigned short* xb  = (unsigned short*)alloc((size_t)B_ * T_ * D_ * 2);
    unsigned short* Wt  = (unsigned short*)alloc((size_t)H_ * D_ * 2);
    unsigned short* Ut  = (unsigned short*)alloc((size_t)H_ * H_ * 2);
    unsigned short* Wdt = (unsigned short*)alloc((size_t)V_ * H_ * 2);
    float*          xw  = (float*)alloc((size_t)T_ * B_ * H_ * 4);
    unsigned short* h0  = (unsigned short*)alloc((size_t)B_ * H_ * 2);
    unsigned short* h1  = (unsigned short*)alloc((size_t)B_ * H_ * 2);
    float*      logits  = (float*)alloc((size_t)B_ * V_ * 4);
    unsigned int*  cnt  = (unsigned int*)alloc(256);

    // bf16 conversion / weight transposition
    f32_to_bf16_kernel<<<(B_ * T_ * D_ + 255) / 256, 256, 0, stream>>>(x, xb, B_ * T_ * D_);
    transpose_to_bf16_kernel<<<(D_ * H_ + 255) / 256, 256, 0, stream>>>(W, Wt, D_, H_);
    transpose_to_bf16_kernel<<<(H_ * H_ + 255) / 256, 256, 0, stream>>>(U, Ut, H_, H_);
    transpose_to_bf16_kernel<<<(H_ * V_ + 255) / 256, 256, 0, stream>>>(Wd, Wdt, H_, V_);
    zero_u16_kernel<<<(B_ * H_ + 255) / 256, 256, 0, stream>>>(h0, B_ * H_);
    zero_u32_kernel<<<1, 32, 0, stream>>>(cnt);

    // xW + b for all timesteps: 131072 tiles, 4 waves/WG
    proj_gemm_kernel<<<(2048 * 64) / 4, 128, 0, stream>>>(xb, Wt, bv, xw);

    // Whole recurrence in ONE persistent kernel: 64 resident WGs, U held in
    // LDS for all 512 steps, atomic-counter step barrier.
    rnn_scan_kernel<<<64, 128, 0, stream>>>(h0, h1, Ut, xw, cnt);
    // after t = 511 the final h lives in h0

    logits_gemm_kernel<<<(4 * 2000) / 4, 128, 0, stream>>>(h0, Wdt, bd, logits);
    softmax_kernel<<<B_, 256, 0, stream>>>(logits, out);
}
